// RandProjector_19885698580808
// MI455X (gfx1250) — matrix operand order repaired and checked by reading, not compiled, not run on hardware
//
#include <hip/hip_runtime.h>
#include <math.h>

// ---------------- problem constants ----------------
#define S_ROWS 16384
#define IN_DIM 1024
#define NPROJ  100
#define PDIM   64
#define DCOLS  (NPROJ * PDIM)   // 6400
#define BINS   20

// ---------------- GEMM tiling ----------------
#define BM 128      // rows per block
#define BN 128      // cols per block
#define BK 64       // K per stage (2 WMMA K-subs)
#define BKP 72      // padded LDS row (ushorts) -> 144B stride, 16B aligned, conflict-free

typedef __attribute__((ext_vector_type(16))) __bf16 v16bf;
typedef __attribute__((ext_vector_type(8)))  float  v8f;

union Frag16 { v16bf v; uint4 q[2]; };
union Acc    { v8f   v; float f[8]; };

// -------- round-to-nearest-even f32 -> bf16 (finite inputs) --------
__device__ __forceinline__ unsigned short f32_to_bf16(float f) {
  unsigned u = __float_as_uint(f);
  u += 0x7FFFu + ((u >> 16) & 1u);
  return (unsigned short)(u >> 16);
}

__global__ __launch_bounds__(256)
void convert_bf16_kernel(const float* __restrict__ in,
                         unsigned short* __restrict__ out, int n4) {
  int i = blockIdx.x * 256 + threadIdx.x;
  if (i < n4) {
    float4 f = ((const float4*)in)[i];
    ushort4 o;
    o.x = f32_to_bf16(f.x);
    o.y = f32_to_bf16(f.y);
    o.z = f32_to_bf16(f.z);
    o.w = f32_to_bf16(f.w);
    ((ushort4*)out)[i] = o;
  }
}

// -------- fused bf16-WMMA GEMM + per-column histogram --------
// grid: (DCOLS/BN, S_ROWS/BM) ; block: 256 threads = 8 waves (wave32)
// wave tile: 32 rows x 64 cols = 2x4 grid of 16x16 accumulators
__global__ __launch_bounds__(256)
void proj_hist_kernel(const unsigned short* __restrict__ Xb,   // [S_ROWS][IN_DIM] bf16
                      const unsigned short* __restrict__ Wb,   // [DCOLS][IN_DIM] bf16
                      const float* __restrict__ mins,
                      const float* __restrict__ maxs,
                      float* __restrict__ ghist)               // [DCOLS][BINS]
{
  __shared__ unsigned short As[BM][BKP];     // 18.0 KB
  __shared__ unsigned short Bs[BN][BKP];     // 18.0 KB
  __shared__ float          hist[BN * BINS]; // 10.0 KB

  const int tid  = threadIdx.x;
  const int lane = tid & 31;
  const int w    = tid >> 5;
  const int wm   = w & 3;      // 0..3 -> 32-row strip
  const int wn   = w >> 2;     // 0..1 -> 64-col strip
  const int r    = lane & 15;  // row/col within 16x16 tile
  const int hi   = lane >> 4;  // lane half

  for (int i = tid; i < BN * BINS; i += 256) hist[i] = 0.0f;

  Acc acc[2][4];
#pragma unroll
  for (int tm = 0; tm < 2; ++tm)
#pragma unroll
    for (int tn = 0; tn < 4; ++tn)
#pragma unroll
      for (int v = 0; v < 8; ++v) acc[tm][tn].f[v] = 0.0f;

  // per-thread staging coordinates: 1024 x 16B chunks per tile, 4 per thread
  int srow[4], sk8[4];
#pragma unroll
  for (int c = 0; c < 4; ++c) {
    int idx = tid + c * 256;
    srow[c] = idx >> 3;
    sk8[c]  = (idx & 7) * 8;
  }

  const size_t aBase = (size_t)(blockIdx.y) * BM * IN_DIM;
  const size_t bBase = (size_t)(blockIdx.x) * BN * IN_DIM;

  for (int kk = 0; kk < IN_DIM; kk += BK) {
    __syncthreads();
    // ---- stage A (128x64) and B (128x64), 8 x 16B chunks per thread ----
#pragma unroll
    for (int c = 0; c < 4; ++c) {
      *(uint4*)&As[srow[c]][sk8[c]] =
          *(const uint4*)&Xb[aBase + (size_t)srow[c] * IN_DIM + kk + sk8[c]];
      *(uint4*)&Bs[srow[c]][sk8[c]] =
          *(const uint4*)&Wb[bBase + (size_t)srow[c] * IN_DIM + kk + sk8[c]];
    }
    __syncthreads();

    // ---- 2 K-subs of 32; per sub: 6 fragments, 8 WMMAs ----
#pragma unroll
    for (int s = 0; s < 2; ++s) {
      const int ks = s * 32;
      Frag16 a[2], b[4];
#pragma unroll
      for (int tm = 0; tm < 2; ++tm) {
        int row = wm * 32 + tm * 16 + r;
        a[tm].q[0] = *(const uint4*)&As[row][ks + hi * 8];       // K {0..7}|{8..15}
        a[tm].q[1] = *(const uint4*)&As[row][ks + hi * 8 + 16];  // K {16..23}|{24..31}
      }
#pragma unroll
      for (int tn = 0; tn < 4; ++tn) {
        int row = wn * 64 + tn * 16 + r;
        b[tn].q[0] = *(const uint4*)&Bs[row][ks + hi * 8];       // K {0..7}|{8..15}
        b[tn].q[1] = *(const uint4*)&Bs[row][ks + hi * 8 + 16];  // K {16..23}|{24..31}
      }
#pragma unroll
      for (int tm = 0; tm < 2; ++tm)
#pragma unroll
        for (int tn = 0; tn < 4; ++tn)
          acc[tm][tn].v = __builtin_amdgcn_wmma_f32_16x16x32_bf16(
              false, a[tm].v, false, b[tn].v, (short)0, acc[tm][tn].v,
              false, false);
    }
  }

  // ---- bin the 64 accumulated proj values this lane owns ----
#pragma unroll
  for (int tn = 0; tn < 4; ++tn) {
    const int colL = wn * 64 + tn * 16 + r;            // C/D layout: N = lane&15 (+tile)
    const int d    = blockIdx.x * BN + colL;
    const float mn = mins[d];
    const float mx = maxs[d];
    const float invw = (float)BINS / (mx - mn);
#pragma unroll
    for (int tm = 0; tm < 2; ++tm) {
#pragma unroll
      for (int v = 0; v < 8; ++v) {
        float val = acc[tm][tn].f[v];
        if (val >= mn && val <= mx) {
          int b = (int)floorf((val - mn) * invw);
          b = b < 0 ? 0 : (b > BINS - 1 ? BINS - 1 : b);
          atomicAdd(&hist[colL * BINS + b], 1.0f);     // ds_add_f32
        }
      }
    }
  }
  __syncthreads();

  // ---- flush block histogram to global ----
  const size_t gbase = (size_t)blockIdx.x * BN * BINS;
  for (int i = tid; i < BN * BINS; i += 256)
    atomicAdd(&ghist[gbase + i], hist[i]);             // global_atomic_add_f32
}

// -------- L2 normalize each (proj, dim) group of BINS --------
__global__ __launch_bounds__(256)
void normalize_kernel(const float* __restrict__ ghist, float* __restrict__ out) {
  int g = blockIdx.x * 256 + threadIdx.x;
  if (g < DCOLS) {
    float h[BINS];
    float ss = 0.0f;
#pragma unroll
    for (int j = 0; j < BINS; ++j) {
      h[j] = ghist[g * BINS + j];
      ss += h[j] * h[j];
    }
    float denom = fmaxf(sqrtf(ss), 1e-12f);
    float inv = 1.0f / denom;
#pragma unroll
    for (int j = 0; j < BINS; ++j) out[g * BINS + j] = h[j] * inv;
  }
}

extern "C" void kernel_launch(void* const* d_in, const int* in_sizes, int n_in,
                              void* d_out, int out_size, void* d_ws, size_t ws_size,
                              hipStream_t stream) {
  const float* x    = (const float*)d_in[0];   // [16384,1024]
  const float* W    = (const float*)d_in[1];   // [6400,1024]
  const float* mins = (const float*)d_in[2];   // [6400]
  const float* maxs = (const float*)d_in[3];   // [6400]
  float* out = (float*)d_out;                  // [100,64,20]

  unsigned short* Xb = (unsigned short*)d_ws;                 // 32 MB
  unsigned short* Wb = Xb + (size_t)S_ROWS * IN_DIM;          // 12.8 MB
  float* ghist = (float*)(Wb + (size_t)DCOLS * IN_DIM);       // 512 KB

  hipMemsetAsync(ghist, 0, (size_t)DCOLS * BINS * sizeof(float), stream);

  const int n4x = S_ROWS * IN_DIM / 4;   // 4,194,304
  convert_bf16_kernel<<<(n4x + 255) / 256, 256, 0, stream>>>(x, Xb, n4x);
  const int n4w = DCOLS * IN_DIM / 4;    // 1,638,400
  convert_bf16_kernel<<<(n4w + 255) / 256, 256, 0, stream>>>(W, Wb, n4w);

  dim3 grid(DCOLS / BN, S_ROWS / BM);    // (50, 128)
  proj_hist_kernel<<<grid, 256, 0, stream>>>(Xb, Wb, mins, maxs, ghist);

  normalize_kernel<<<(DCOLS + 255) / 256, 256, 0, stream>>>(ghist, out);
}
